// TreeTLSTMQFunction_62560493634054
// MI455X (gfx1250) — matrix operand order repaired and checked by reading, not compiled, not run on hardware
//
#include <hip/hip_runtime.h>

// ---------------------------------------------------------------------------
// TreeLSTM Q-function for MI455X (gfx1250), wave32, bf16 WMMA everywhere.
//  B=8192, L=16, S=256, T=1000, C=500, z-width = 5*S = 1280, Kcat = 3*S = 768.
// ---------------------------------------------------------------------------

typedef __bf16 bf16;
typedef __attribute__((ext_vector_type(16))) __bf16 v16bf;
typedef __attribute__((ext_vector_type(8)))  __bf16 v8bf;
typedef __attribute__((ext_vector_type(8)))  float  v8f;

union FragBF { v16bf v; v8bf h[2]; };

#define B_   8192
#define L_   16
#define S_   256
#define NZ   1280   // 5*S
#define KC   768    // 3*S
#define KJC  512    // 500 padded to 512

__device__ __forceinline__ float sigm(float x) { return 1.0f / (1.0f + expf(-x)); }

// ---------------------------------------------------------------------------
// Prep: transpose weights to N-major bf16 (B-fragment layout: each WMMA lane
// reads two 16B chunks of K), fold the three z-biases into one.
// ---------------------------------------------------------------------------
__global__ void prep_kernel(const float* __restrict__ W1, const float* __restrict__ W2,
                            const float* __restrict__ W0,
                            const float* __restrict__ b1, const float* __restrict__ b2,
                            const float* __restrict__ b0,
                            const float* __restrict__ jcW, const float* __restrict__ fcW,
                            bf16* __restrict__ Wt, float* __restrict__ biasz,
                            bf16* __restrict__ jcWt, bf16* __restrict__ fcWt)
{
  const int i = blockIdx.x * 256 + threadIdx.x;
  if (i < NZ * KC) {                       // Wt[nc][k], k: 0..255=W1, 256..511=W2, 512..767=W0
    const int nc = i / KC, k = i % KC;
    float v;
    if (k < 256)      v = W1[(size_t)k * NZ + nc];
    else if (k < 512) v = W2[(size_t)(k - 256) * NZ + nc];
    else              v = W0[(size_t)(k - 512) * NZ + nc];
    Wt[i] = (bf16)v;
  }
  if (i < NZ) biasz[i] = b1[i] + b2[i] + b0[i];
  if (i < 256 * KJC) {                     // jcWt[nc][k] (zero-pad k>=500), fcWt[nc][k]
    const int nc = i >> 9, k = i & 511;
    jcWt[i] = (k < 500) ? (bf16)jcW[(size_t)k * 256 + nc] : (bf16)0.0f;
    fcWt[i] = (bf16)fcW[(size_t)k * 256 + nc];
  }
}

// ---------------------------------------------------------------------------
// Gather + LayerNorm -> bf16 rows. One wave per row (8 rows / 256-thr block).
// Rows [0, B*L): leaves -> hleaf ; rows [B*L, B*L+B): actions -> hact.
// ---------------------------------------------------------------------------
__global__ void embed_ln_kernel(const int* __restrict__ leaf_ids,
                                const int* __restrict__ action_ids,
                                const float* __restrict__ emb,
                                const float* __restrict__ ln_g,
                                const float* __restrict__ ln_b,
                                bf16* __restrict__ hleaf, bf16* __restrict__ hact)
{
  const int row  = blockIdx.x * 8 + (threadIdx.x >> 5);
  const int lane = threadIdx.x & 31;
  int id; bf16* dst;
  if (row < B_ * L_) { id = leaf_ids[row]; dst = hleaf + (size_t)row * S_; }
  else { const int r = row - B_ * L_; id = action_ids[r]; dst = hact + (size_t)r * S_; }
  const float* e = emb + (size_t)id * S_;
  float v[8], s = 0.0f, sq = 0.0f;
#pragma unroll
  for (int i = 0; i < 8; ++i) { float x = e[i * 32 + lane]; v[i] = x; s += x; sq += x * x; }
#pragma unroll
  for (int m = 16; m >= 1; m >>= 1) { s += __shfl_xor(s, m, 32); sq += __shfl_xor(sq, m, 32); }
  const float mu  = s * (1.0f / S_);
  const float var = sq * (1.0f / S_) - mu * mu;
  const float rs  = rsqrtf(var + 1e-5f);
#pragma unroll
  for (int i = 0; i < 8; ++i) {
    const int k = i * 32 + lane;
    dst[k] = (bf16)((v[i] - mu) * rs * ln_g[k] + ln_b[k]);
  }
}

// ---------------------------------------------------------------------------
// X = join_cols(B*15, 500) @ jc_W + jc_b  -> bf16 (B*15, 256).
// WMMA GEMM: block tile 32(M) x 256(N), K=512 (padded), 512 thr = 16 waves.
// Wave (mi, wi): M half mi*16, two 16-col tiles at wi*32.
// ---------------------------------------------------------------------------
__global__ void jc_gemm_kernel(const float* __restrict__ jc,
                               const bf16* __restrict__ jcWt,
                               const float* __restrict__ jc_b,
                               bf16* __restrict__ X)
{
  __shared__ __align__(16) bf16 Alds[32][KJC + 8];   // +8 bf16 pad -> bank-conflict free
  const int rowBase = blockIdx.x * 32;
  const int t = threadIdx.x;
  // stage A: f32 -> bf16 (row stride 500 f32 = 2000B, 16B aligned -> float4)
  for (int i = t; i < 32 * 125; i += 512) {
    const int r = i / 125, seg = i % 125;
    const float4 f = ((const float4*)(jc + (size_t)(rowBase + r) * 500))[seg];
    bf16* d = &Alds[r][seg * 4];
    d[0] = (bf16)f.x; d[1] = (bf16)f.y; d[2] = (bf16)f.z; d[3] = (bf16)f.w;
  }
  for (int i = t; i < 32 * 12; i += 512) Alds[i / 12][500 + i % 12] = (bf16)0.0f;
  __syncthreads();

  const int w = t >> 5, lane = t & 31;
  const int mi = w >> 3, wi = w & 7;
  const int lm = lane & 15, half = lane >> 4, mt = mi * 16;

  v8f acc[2] = {};
  const bf16* wr[2];
#pragma unroll
  for (int st = 0; st < 2; ++st) wr[st] = jcWt + (size_t)((wi * 2 + st) * 16 + lm) * KJC;

  for (int kb = 0; kb < KJC / 32; ++kb) {
    FragBF a;
    a.h[0] = *(const v8bf*)&Alds[mt + lm][kb * 32 + half * 8];
    a.h[1] = *(const v8bf*)&Alds[mt + lm][kb * 32 + 16 + half * 8];
#pragma unroll
    for (int st = 0; st < 2; ++st) {
      FragBF bb;
      const bf16* p = wr[st] + kb * 32 + half * 8;
      bb.h[0] = *(const v8bf*)p;
      bb.h[1] = *(const v8bf*)(p + 16);
      acc[st] = __builtin_amdgcn_wmma_f32_16x16x32_bf16(false, a.v, false, bb.v,
                                                        (short)0, acc[st], false, false);
    }
  }
#pragma unroll
  for (int st = 0; st < 2; ++st) {
    const int col = (wi * 2 + st) * 16 + lm;
    const float bias = jc_b[col];
#pragma unroll
    for (int r = 0; r < 8; ++r) {
      const int row = rowBase + mt + r + 8 * half;
      X[(size_t)row * S_ + col] = (bf16)(acc[st][r] + bias);
    }
  }
}

// ---------------------------------------------------------------------------
// One tree level, fully fused: z = [hl|hr|x] @ Wcat + biasz -> gates -> c,h.
// Block tile 32(M) x 1280(N), K=768, 512 thr = 16 waves.
// Wave (mi, wi) owns s-strip [wi*32, wi*32+32) across ALL FIVE gates, so the
// five gate values of (row, s) live in the same lane/accumulator slot:
// the whole LSTM cell update happens in registers.
// ---------------------------------------------------------------------------
__global__ void tree_level_kernel(const bf16* __restrict__ h_in,   // (B, 2n, S) bf16
                                  const float* __restrict__ c_in,  // (B, 2n, S) f32 (or null)
                                  const bf16* __restrict__ X,      // (B*15, S) bf16
                                  const bf16* __restrict__ Wt,     // (1280, 768) bf16
                                  const float* __restrict__ biasz, // (1280) f32
                                  bf16* __restrict__ h_out,        // (B, n, S)
                                  float* __restrict__ c_out,       // (B, n, S)
                                  int n, int ln2n, int off, int zero_c)
{
  __shared__ __align__(16) bf16 Alds[32][KC + 8];
  const int rowBase = blockIdx.x * 32;
  const int t = threadIdx.x;

  // Stage A rows: K 0..511 = (hl|hr) which are CONTIGUOUS in h_in; 512..767 = x_node.
  for (int i = t; i < 32 * 96; i += 512) {
    const int r = i / 96, seg = i % 96;
    const int row = rowBase + r;
    const int b = row >> ln2n;
    const int j = row & (n - 1);
    const v8bf* src;
    if (seg < 64) {
      const bf16* hp = h_in + (size_t)((b << (ln2n + 1)) + 2 * j) * S_;
      src = (const v8bf*)(hp + seg * 8);
    } else {
      const bf16* xp = X + (size_t)(b * 15 + off + j) * S_;
      src = (const v8bf*)(xp + (seg - 64) * 8);
    }
    *(v8bf*)&Alds[r][seg * 8] = *src;
  }
  __syncthreads();

  const int w = t >> 5, lane = t & 31;
  const int mi = w >> 3, wi = w & 7;
  const int lm = lane & 15, half = lane >> 4, mt = mi * 16;

  v8f acc[2][5] = {};
  const bf16* wr[2][5];
#pragma unroll
  for (int st = 0; st < 2; ++st)
#pragma unroll
    for (int g = 0; g < 5; ++g)
      wr[st][g] = Wt + (size_t)(g * 256 + wi * 32 + st * 16 + lm) * KC;

  for (int kb = 0; kb < KC / 32; ++kb) {
    FragBF a;
    a.h[0] = *(const v8bf*)&Alds[mt + lm][kb * 32 + half * 8];
    a.h[1] = *(const v8bf*)&Alds[mt + lm][kb * 32 + 16 + half * 8];
#pragma unroll
    for (int st = 0; st < 2; ++st) {
#pragma unroll
      for (int g = 0; g < 5; ++g) {
        FragBF bb;
        const bf16* p = wr[st][g] + kb * 32 + half * 8;
        bb.h[0] = *(const v8bf*)p;
        bb.h[1] = *(const v8bf*)(p + 16);
        acc[st][g] = __builtin_amdgcn_wmma_f32_16x16x32_bf16(false, a.v, false, bb.v,
                                                             (short)0, acc[st][g], false, false);
      }
    }
  }

  // Gate epilogue, entirely in registers.
#pragma unroll
  for (int st = 0; st < 2; ++st) {
    const int s = wi * 32 + st * 16 + lm;
    const float bza = biasz[s];
    const float bzi = biasz[256 + s];
    const float bz1 = biasz[512 + s];
    const float bz2 = biasz[768 + s];
    const float bzo = biasz[1024 + s];
#pragma unroll
    for (int r = 0; r < 8; ++r) {
      const int row = rowBase + mt + r + 8 * half;
      const int b = row >> ln2n;
      const int j = row & (n - 1);
      float cl = 0.0f, cr = 0.0f;
      if (!zero_c) {
        const size_t cb = (size_t)((b << (ln2n + 1)) + 2 * j) * S_ + s;
        cl = c_in[cb];
        cr = c_in[cb + S_];
      }
      const float av = acc[st][0][r] + bza;
      const float iv = acc[st][1][r] + bzi;
      const float f1 = acc[st][2][r] + bz1;
      const float f2 = acc[st][3][r] + bz2;
      const float ov = acc[st][4][r] + bzo;
      const float cc = tanhf(av) * sigm(iv) + sigm(f1) * cl + sigm(f2) * cr;
      const float hh = sigm(ov) * tanhf(cc);
      const size_t ob = (size_t)row * S_ + s;
      c_out[ob] = cc;
      h_out[ob] = (bf16)hh;
    }
  }
}

// ---------------------------------------------------------------------------
// Head: out[b] = relu([h_root|action_h] @ fc_W + fc_b) @ out_W + out_b.
// GEMM 32x256, K=512 bf16 WMMA, fused relu + out_W dot via LDS reduction.
// ---------------------------------------------------------------------------
__global__ void fc_out_kernel(const bf16* __restrict__ hroot, const bf16* __restrict__ hact,
                              const bf16* __restrict__ fcWt, const float* __restrict__ fc_b,
                              const float* __restrict__ out_W, const float* __restrict__ out_b,
                              float* __restrict__ out)
{
  __shared__ __align__(16) bf16 Alds[32][KJC + 8];
  __shared__ float rowsum[32];
  const int rowBase = blockIdx.x * 32;
  const int t = threadIdx.x;
  if (t < 32) rowsum[t] = 0.0f;
  for (int i = t; i < 32 * 64; i += 512) {
    const int r = i >> 6, seg = i & 63;
    const v8bf* src = (seg < 32)
        ? (const v8bf*)(hroot + (size_t)(rowBase + r) * S_ + seg * 8)
        : (const v8bf*)(hact  + (size_t)(rowBase + r) * S_ + (seg - 32) * 8);
    *(v8bf*)&Alds[r][seg * 8] = *src;
  }
  __syncthreads();

  const int w = t >> 5, lane = t & 31;
  const int mi = w >> 3, wi = w & 7;
  const int lm = lane & 15, half = lane >> 4, mt = mi * 16;

  v8f acc[2] = {};
  const bf16* wr[2];
#pragma unroll
  for (int st = 0; st < 2; ++st) wr[st] = fcWt + (size_t)((wi * 2 + st) * 16 + lm) * KJC;

  for (int kb = 0; kb < KJC / 32; ++kb) {
    FragBF a;
    a.h[0] = *(const v8bf*)&Alds[mt + lm][kb * 32 + half * 8];
    a.h[1] = *(const v8bf*)&Alds[mt + lm][kb * 32 + 16 + half * 8];
#pragma unroll
    for (int st = 0; st < 2; ++st) {
      FragBF bb;
      const bf16* p = wr[st] + kb * 32 + half * 8;
      bb.h[0] = *(const v8bf*)p;
      bb.h[1] = *(const v8bf*)(p + 16);
      acc[st] = __builtin_amdgcn_wmma_f32_16x16x32_bf16(false, a.v, false, bb.v,
                                                        (short)0, acc[st], false, false);
    }
  }

  float part[8] = {0.f, 0.f, 0.f, 0.f, 0.f, 0.f, 0.f, 0.f};
#pragma unroll
  for (int st = 0; st < 2; ++st) {
    const int col = (wi * 2 + st) * 16 + lm;
    const float bias = fc_b[col];
    const float ow = out_W[col];
#pragma unroll
    for (int r = 0; r < 8; ++r) {
      const float y = fmaxf(acc[st][r] + bias, 0.0f);
      part[r] += y * ow;
    }
  }
#pragma unroll
  for (int r = 0; r < 8; ++r) atomicAdd(&rowsum[mt + r + 8 * half], part[r]);
  __syncthreads();
  if (t < 32) out[rowBase + t] = rowsum[t] + out_b[0];
}

// ---------------------------------------------------------------------------
extern "C" void kernel_launch(void* const* d_in, const int* in_sizes, int n_in,
                              void* d_out, int out_size, void* d_ws, size_t ws_size,
                              hipStream_t stream) {
  const int*   leaf_ids   = (const int*)  d_in[0];
  const int*   action_ids = (const int*)  d_in[1];
  const float* join_cols  = (const float*)d_in[2];
  const float* emb        = (const float*)d_in[3];
  const float* ln_g       = (const float*)d_in[4];
  const float* ln_b       = (const float*)d_in[5];
  const float* jc_W       = (const float*)d_in[6];
  const float* jc_b       = (const float*)d_in[7];
  const float* W1         = (const float*)d_in[8];
  const float* b1         = (const float*)d_in[9];
  const float* W2         = (const float*)d_in[10];
  const float* b2         = (const float*)d_in[11];
  const float* W0         = (const float*)d_in[12];
  const float* b0         = (const float*)d_in[13];
  const float* fc_W       = (const float*)d_in[14];
  const float* fc_b       = (const float*)d_in[15];
  const float* out_W      = (const float*)d_in[16];
  const float* out_b      = (const float*)d_in[17];
  float* out = (float*)d_out;

  char* ws = (char*)d_ws;
  size_t cur = 0;
  auto alloc = [&](size_t bytes) -> void* {
    void* p = ws + cur;
    cur = (cur + bytes + 255) & ~(size_t)255;
    return p;
  };
  bf16*  Wt    = (bf16*) alloc((size_t)NZ * KC * 2);        //  1.97 MB
  bf16*  jcWt  = (bf16*) alloc((size_t)256 * KJC * 2);      //  0.26 MB
  bf16*  fcWt  = (bf16*) alloc((size_t)256 * KJC * 2);      //  0.26 MB
  float* biasz = (float*)alloc((size_t)NZ * 4);
  bf16*  X     = (bf16*) alloc((size_t)B_ * 15 * S_ * 2);   // 62.9 MB
  bf16*  hA    = (bf16*) alloc((size_t)B_ * 16 * S_ * 2);   // 67.1 MB
  bf16*  hB    = (bf16*) alloc((size_t)B_ * 8  * S_ * 2);   // 33.6 MB
  float* cA    = (float*)alloc((size_t)B_ * 8  * S_ * 4);   // 67.1 MB
  float* cB    = (float*)alloc((size_t)B_ * 4  * S_ * 4);   // 33.6 MB
  bf16*  actH  = (bf16*) alloc((size_t)B_ * S_ * 2);        //  4.2 MB
  (void)ws_size; (void)in_sizes; (void)n_in; (void)out_size;

  // 1) weight transposes + folded bias
  prep_kernel<<<(NZ * KC + 255) / 256, 256, 0, stream>>>(W1, W2, W0, b1, b2, b0,
                                                         jc_W, fc_W, Wt, biasz, jcWt, fcWt);
  // 2) gather + layernorm (B*16 leaf rows + B action rows; 8 rows per block, exact)
  embed_ln_kernel<<<(B_ * L_ + B_) / 8, 256, 0, stream>>>(leaf_ids, action_ids, emb,
                                                          ln_g, ln_b, hA, actH);
  // 3) X = join_cols @ jc_W + jc_b for all 15 nodes (M = B*15 = 122880)
  jc_gemm_kernel<<<B_ * 15 / 32, 512, 0, stream>>>(join_cols, jcWt, jc_b, X);
  // 4) four tree levels, ping-ponging h (bf16) and c (f32)
  tree_level_kernel<<<B_ * 8 / 32, 512, 0, stream>>>(hA, nullptr, X, Wt, biasz, hB, cA, 8, 3, 0, 1);
  tree_level_kernel<<<B_ * 4 / 32, 512, 0, stream>>>(hB, cA, X, Wt, biasz, hA, cB, 4, 2, 8, 0);
  tree_level_kernel<<<B_ * 2 / 32, 512, 0, stream>>>(hA, cB, X, Wt, biasz, hB, cA, 2, 1, 12, 0);
  tree_level_kernel<<<B_ * 1 / 32, 512, 0, stream>>>(hB, cA, X, Wt, biasz, hA, cB, 1, 0, 14, 0);
  // 5) head (root h is in hA, B rows)
  fc_out_kernel<<<B_ / 32, 512, 0, stream>>>(hA, actH, fcWt, fc_b, out_W, out_b, out);
}
